// RealTimeTransformer_33182917328974
// MI455X (gfx1250) — matrix operand order repaired and checked by reading, not compiled, not run on hardware
//
#include <hip/hip_runtime.h>

// ---------------------------------------------------------------------------
// Transformer encoder layer for MI455X (gfx1250, wave32, WMMA).
// B=4, S=2048, H=1024, NH=8, D=128. All GEMMs on v_wmma_f32_16x16x32_bf16.
// GEMM operand staging: A via global_load_async_to_lds_b128 (ASYNCcnt),
// B via Tensor Data Mover tensor_load_to_lds (TENSORcnt) when available.
// ---------------------------------------------------------------------------

constexpr int kB  = 4;
constexpr int kS  = 2048;
constexpr int kH  = 1024;
constexpr int kNH = 8;
constexpr int kD  = 128;          // head dim
constexpr int kBS = kB * kS;      // 8192 tokens

typedef __attribute__((ext_vector_type(16))) __bf16        v16bf;
typedef __attribute__((ext_vector_type(8)))  float         v8f;
typedef __attribute__((ext_vector_type(4)))  unsigned int  v4u;
typedef __attribute__((ext_vector_type(8)))  unsigned int  v8u;
typedef __attribute__((ext_vector_type(4)))  unsigned int  u32x4;
typedef __attribute__((ext_vector_type(4)))  int           i32x4;
typedef __attribute__((ext_vector_type(8)))  int           i32x8;

__device__ inline unsigned short f2bf(float f) {
  unsigned u = __builtin_bit_cast(unsigned, f);
  unsigned r = u + 0x7FFFu + ((u >> 16) & 1u);   // round-to-nearest-even
  return (unsigned short)(r >> 16);
}

__device__ inline v8f zero8() {
  v8f z = {0.f, 0.f, 0.f, 0.f, 0.f, 0.f, 0.f, 0.f};
  return z;
}

// A-fragment (16x32 bf16, MxK): lane l<16 -> row l, K in {hi*8..} u {16+hi*8..}
__device__ inline v16bf load_a_frag(const unsigned short* t, int ldk, int lane) {
  const int l = lane & 15, hi = lane >> 4;
  const unsigned short* r = t + (size_t)l * ldk + hi * 8;
  v4u lo = *(const v4u*)(r);
  v4u hh = *(const v4u*)(r + 16);
  v8u u  = __builtin_shufflevector(lo, hh, 0, 1, 2, 3, 4, 5, 6, 7);
  return __builtin_bit_cast(v16bf, u);
}

// B-fragment (32x16 bf16, KxN) read from B^T stored row-major (16 rows x ldk)
__device__ inline v16bf load_b_frag(const unsigned short* t, int ldk, int lane) {
  const int n = lane & 15, hi = lane >> 4;
  const unsigned short* r = t + (size_t)n * ldk + hi * 8;
  v4u lo = *(const v4u*)(r);
  v4u hh = *(const v4u*)(r + 16);
  v8u u  = __builtin_shufflevector(lo, hh, 0, 1, 2, 3, 4, 5, 6, 7);
  return __builtin_bit_cast(v16bf, u);
}

__device__ inline v8f wmma_bf16(v16bf a, v16bf b, v8f c) {
  return __builtin_amdgcn_wmma_f32_16x16x32_bf16(false, a, false, b, (short)0, c,
                                                 false, false);
}

__device__ inline float red_max16(float v) {
#pragma unroll
  for (int m = 1; m < 16; m <<= 1) v = fmaxf(v, __shfl_xor(v, m, 32));
  return v;
}
__device__ inline float red_sum16(float v) {
#pragma unroll
  for (int m = 1; m < 16; m <<= 1) v += __shfl_xor(v, m, 32);
  return v;
}

// ---------------------------------------------------------------------------
// CDNA5 async/TDM data movers
// ---------------------------------------------------------------------------
// Per-lane async copy of 16B global -> LDS (tracked by ASYNCcnt).
// ISA: GLOBAL_LOAD_ASYNC_TO_LDS_B128 vdst(LDS byte addr), vaddr(64b), off
__device__ inline void async_g2l_b128(unsigned lds_off, const void* gaddr) {
  asm volatile("global_load_async_to_lds_b128 %0, %1, off"
               :
               : "v"(lds_off), "v"((unsigned long long)(size_t)gaddr)
               : "memory");
}

__device__ inline void wait_asynccnt0() {
#if __has_builtin(__builtin_amdgcn_s_wait_asynccnt)
  __builtin_amdgcn_s_wait_asynccnt(0);
#else
  asm volatile("s_wait_asynccnt 0" ::: "memory");
#endif
}

#if __has_builtin(__builtin_amdgcn_tensor_load_to_lds)
#define HAVE_TDM 1
// 2D tile load: 64 rows x 32 bf16, row stride K elements, via Tensor Data Mover.
// D# per CDNA5 ISA (group0 + group1, groups 2/3 zero).
__device__ inline void tdm_load_64x32_bf16(unsigned lds_off, const void* gptr,
                                           unsigned Kel) {
  const unsigned long long ga = (unsigned long long)(size_t)gptr;
  u32x4 g0 = {
      1u,                                                   // count=1 (valid D#)
      lds_off,                                              // lds_addr
      (unsigned)ga,                                         // global_addr[31:0]
      ((unsigned)(ga >> 32) & 0x01FFFFFFu) | (2u << 30)     // ga[56:32] | type=2
  };
  i32x8 g1 = {
      (int)(1u << 16),                        // data_size=1 (2 bytes/elem)
      (int)((Kel & 0xFFFFu) << 16),           // tensor_dim0[15:0]  (bits 63:48)
      (int)((Kel >> 16) | (64u << 16)),       // tensor_dim0[31:16] | tensor_dim1=64
      (int)(32u << 16),                       // tile_dim0=32 (bits 127:112)
      (int)64,                                // tile_dim1=64, tile_dim2=0
      (int)Kel,                               // tensor_dim0_stride[31:0]
      0,                                      // stride0[47:32] | stride1[15:0]
      0
  };
  i32x4 gz = {0, 0, 0, 0};
#if __clang_major__ >= 23
  i32x8 gz8 = {0, 0, 0, 0, 0, 0, 0, 0};
  __builtin_amdgcn_tensor_load_to_lds(g0, g1, gz, gz, gz8, 0);
#else
  __builtin_amdgcn_tensor_load_to_lds(g0, g1, gz, gz, 0);
#endif
}
#endif

__device__ inline unsigned lds_offset_of(const void* p) {
  return (unsigned)(size_t)p;  // low 32 bits of flat shared addr = LDS byte addr
}

// ---------------------------------------------------------------------------
// fp32 -> bf16 conversion
// ---------------------------------------------------------------------------
__global__ void cvt_kernel(const float* __restrict__ src,
                           unsigned short* __restrict__ dst, size_t n) {
  for (size_t i = (size_t)blockIdx.x * blockDim.x + threadIdx.x; i < n;
       i += (size_t)gridDim.x * blockDim.x)
    dst[i] = f2bf(src[i]);
}

// ---------------------------------------------------------------------------
// key padding mask: maskb[token] = (sum|x| == 0) ? -1e30 : 0
// ---------------------------------------------------------------------------
__global__ void __launch_bounds__(256)
mask_kernel(const float* __restrict__ x, float* __restrict__ maskb) {
  const int row = blockIdx.x;
  const float* xr = x + (size_t)row * kH;
  float s = 0.f;
  for (int i = threadIdx.x; i < kH; i += 256) s += fabsf(xr[i]);
#pragma unroll
  for (int m = 1; m < 32; m <<= 1) s += __shfl_xor(s, m, 32);
  __shared__ float red[8];
  if ((threadIdx.x & 31) == 0) red[threadIdx.x >> 5] = s;
  __syncthreads();
  if (threadIdx.x == 0) {
    float t = 0.f;
    for (int w = 0; w < 8; ++w) t += red[w];
    maskb[row] = (t == 0.f) ? -1.0e30f : 0.f;
  }
}

// ---------------------------------------------------------------------------
// GEMM: C[M,N] = A[M,K](bf16) @ W[N,K]^T(bf16) + bias.
// Block tile 128M x 64N, 8 waves, each wave owns a 32x32 output (4 WMMA accs).
// A tile (128x32, 8KB) staged via async global->LDS b128 (2 per thread).
// B tile (64x32, 4KB) staged via TDM tensor_load_to_lds (wave 0 issues).
// EPI: 0 = f32 out, 1 = bf16 out, 2 = relu -> bf16 out.
// ---------------------------------------------------------------------------
template <int EPI>
__global__ void __launch_bounds__(256)
gemm_bf16_kernel(const unsigned short* __restrict__ A,
                 const unsigned short* __restrict__ W,
                 const float* __restrict__ bias,
                 float* __restrict__ outF,
                 unsigned short* __restrict__ outH,
                 int M, int N, int K) {
  __shared__ __align__(16) unsigned short As[128 * 32];
  __shared__ __align__(16) unsigned short Bs[64 * 32];
  const int tid  = threadIdx.x;
  const int lane = tid & 31, wave = tid >> 5;
  const int mt = wave >> 1, ntp = wave & 1;       // 4 M x 2 N wave grid
  const int M0 = blockIdx.y * 128, N0 = blockIdx.x * 64;

  v8f acc[2][2];
#pragma unroll
  for (int i = 0; i < 2; ++i)
#pragma unroll
    for (int j = 0; j < 2; ++j) acc[i][j] = zero8();

  // A-copy mapping: 512 b128 chunks; chunk c -> row c>>2, col (c&3)*8
  const int r0 = tid >> 2, c0 = (tid & 3) * 8;
  const unsigned ldsA = lds_offset_of(&As[0]);
#if !defined(HAVE_TDM)
  const unsigned ldsB = lds_offset_of(&Bs[0]);
#endif

  const int ksteps = K >> 5;
  for (int kb = 0; kb < ksteps; ++kb) {
    const unsigned short* Ak = A + (size_t)M0 * K + kb * 32;
    async_g2l_b128(ldsA + (unsigned)((r0 * 32 + c0) * 2),
                   Ak + (size_t)r0 * K + c0);
    async_g2l_b128(ldsA + (unsigned)(((r0 + 64) * 32 + c0) * 2),
                   Ak + (size_t)(r0 + 64) * K + c0);
#if defined(HAVE_TDM)
    if (wave == 0)
      tdm_load_64x32_bf16(lds_offset_of(&Bs[0]),
                          W + (size_t)N0 * K + kb * 32, (unsigned)K);
#else
    async_g2l_b128(ldsB + (unsigned)((r0 * 32 + c0) * 2),
                   W + (size_t)(N0 + (r0 & 63)) * K + kb * 32 + c0);
#endif
    wait_asynccnt0();
#if defined(HAVE_TDM)
    if (wave == 0) __builtin_amdgcn_s_wait_tensorcnt(0);
#endif
    __syncthreads();

    v16bf a0 = load_a_frag(&As[(mt * 32) * 32], 32, lane);
    v16bf a1 = load_a_frag(&As[(mt * 32 + 16) * 32], 32, lane);
    v16bf b0 = load_b_frag(&Bs[(ntp * 32) * 32], 32, lane);
    v16bf b1 = load_b_frag(&Bs[(ntp * 32 + 16) * 32], 32, lane);
    acc[0][0] = wmma_bf16(a0, b0, acc[0][0]);
    acc[0][1] = wmma_bf16(a0, b1, acc[0][1]);
    acc[1][0] = wmma_bf16(a1, b0, acc[1][0]);
    acc[1][1] = wmma_bf16(a1, b1, acc[1][1]);
    __syncthreads();
  }

#pragma unroll
  for (int j = 0; j < 2; ++j) {
    const int col = N0 + ntp * 32 + j * 16 + (lane & 15);
    const float bv = bias[col];
#pragma unroll
    for (int i = 0; i < 2; ++i)
#pragma unroll
      for (int r = 0; r < 8; ++r) {
        const int row = M0 + mt * 32 + i * 16 + ((lane < 16) ? r : r + 8);
        float v = acc[i][j][r] + bv;
        if (EPI == 2) v = fmaxf(v, 0.f);
        if (EPI == 0) outF[(size_t)row * N + col] = v;
        else          outH[(size_t)row * N + col] = f2bf(v);
      }
  }
}

// ---------------------------------------------------------------------------
// split qkv[B,S,3H] -> q[B,h,S,D], k[B,h,S,D], vT[B,h,D,S]
// ---------------------------------------------------------------------------
__global__ void scatter_qkv_kernel(const unsigned short* __restrict__ qkvh,
                                   unsigned short* __restrict__ qh,
                                   unsigned short* __restrict__ kh,
                                   unsigned short* __restrict__ vth) {
  const size_t total = (size_t)kBS * kH;
  for (size_t idx = (size_t)blockIdx.x * blockDim.x + threadIdx.x; idx < total;
       idx += (size_t)gridDim.x * blockDim.x) {
    const int hd = (int)(idx % kH);
    const int s  = (int)((idx / kH) % kS);
    const int b  = (int)(idx / ((size_t)kH * kS));
    const int h = hd >> 7, dd = hd & 127;
    const size_t src = ((size_t)b * kS + s) * (size_t)(3 * kH);
    const size_t bhs = ((size_t)(b * kNH + h) * kS + s) * kD + dd;
    qh[bhs] = qkvh[src + hd];
    kh[bhs] = qkvh[src + kH + hd];
    vth[((size_t)(b * kNH + h) * kD + dd) * kS + s] = qkvh[src + 2 * kH + hd];
  }
}

// ---------------------------------------------------------------------------
// Flash attention. 8 waves/block; wave owns 16 queries x full d=128.
// ---------------------------------------------------------------------------
__global__ void __launch_bounds__(256)
attn_kernel(const unsigned short* __restrict__ qh,
            const unsigned short* __restrict__ kh,
            const unsigned short* __restrict__ vth,
            const float* __restrict__ maskb,
            unsigned short* __restrict__ ctxh) {
  __shared__ __align__(16) unsigned short Ps[8][16 * 32];
  const int bh = blockIdx.x;
  const int b = bh / kNH, h = bh % kNH;
  const int lane = threadIdx.x & 31, wave = threadIdx.x >> 5;
  const int q0 = blockIdx.y * 128 + wave * 16;
  const float scale = 0.08838834764831845f;      // 1/sqrt(128)

  v16bf qf[4];
  const unsigned short* qbase = qh + ((size_t)bh * kS + q0) * kD;
#pragma unroll
  for (int kc = 0; kc < 4; ++kc) qf[kc] = load_a_frag(qbase + kc * 32, kD, lane);

  float m8[8], l8[8];
  v8f o[8];
#pragma unroll
  for (int r = 0; r < 8; ++r) { m8[r] = -3.0e38f; l8[r] = 0.f; }
#pragma unroll
  for (int dc = 0; dc < 8; ++dc) o[dc] = zero8();

  const float* mrow = maskb + (size_t)b * kS;

  for (int kb = 0; kb < kS; kb += 32) {
    if (kb + 32 < kS) {  // gfx1250 global_prefetch_b8 on next K/V tiles
      __builtin_prefetch(kh + ((size_t)bh * kS + kb + 32) * kD + lane * 8, 0, 1);
      __builtin_prefetch(vth + (size_t)bh * kD * kS + kb + 32 + lane * 4, 0, 1);
    }
    v8f s[2];
#pragma unroll
    for (int half = 0; half < 2; ++half) {
      s[half] = zero8();
      const unsigned short* kbase = kh + ((size_t)bh * kS + kb + half * 16) * kD;
#pragma unroll
      for (int kc = 0; kc < 4; ++kc) {
        v16bf kf = load_b_frag(kbase + kc * 32, kD, lane);
        s[half] = wmma_bf16(qf[kc], kf, s[half]);
      }
      const float mv = mrow[kb + half * 16 + (lane & 15)];
#pragma unroll
      for (int r = 0; r < 8; ++r) s[half][r] = s[half][r] * scale + mv;
    }

    float alpha[8];
#pragma unroll
    for (int r = 0; r < 8; ++r) {
      const float rm = red_max16(fmaxf(s[0][r], s[1][r]));
      const float mn = fmaxf(m8[r], rm);
      alpha[r] = __expf(m8[r] - mn);
      m8[r] = mn;
    }
#pragma unroll
    for (int r = 0; r < 8; ++r) {
      const float p0 = __expf(s[0][r] - m8[r]);
      const float p1 = __expf(s[1][r] - m8[r]);
      s[0][r] = p0;
      s[1][r] = p1;
      l8[r] = l8[r] * alpha[r] + red_sum16(p0 + p1);
    }
#pragma unroll
    for (int dc = 0; dc < 8; ++dc)
#pragma unroll
      for (int r = 0; r < 8; ++r) o[dc][r] *= alpha[r];

    unsigned short* myP = &Ps[wave][0];
#pragma unroll
    for (int r = 0; r < 8; ++r) {
      const int row = (lane < 16) ? r : (r + 8);
      myP[row * 32 + (lane & 15)]      = f2bf(s[0][r]);
      myP[row * 32 + 16 + (lane & 15)] = f2bf(s[1][r]);
    }
    __syncthreads();
    v16bf pf = load_a_frag(myP, 32, lane);
    __syncthreads();

#pragma unroll
    for (int dc = 0; dc < 8; ++dc) {
      v16bf vf = load_b_frag(vth + ((size_t)bh * kD + dc * 16) * kS + kb, kS, lane);
      o[dc] = wmma_bf16(pf, vf, o[dc]);
    }
  }

#pragma unroll
  for (int dc = 0; dc < 8; ++dc)
#pragma unroll
    for (int r = 0; r < 8; ++r) {
      const int row = q0 + ((lane < 16) ? r : (r + 8));
      const int col = h * kD + dc * 16 + (lane & 15);
      ctxh[((size_t)b * kS + row) * (size_t)kH + col] = f2bf(o[dc][r] / l8[r]);
    }
}

// ---------------------------------------------------------------------------
// out = LayerNorm(y + resid) * g + beta;  optional bf16 mirror
// ---------------------------------------------------------------------------
__global__ void __launch_bounds__(256)
add_ln_kernel(const float* __restrict__ y, const float* __restrict__ resid,
              const float* __restrict__ g, const float* __restrict__ beta,
              float* __restrict__ outF, unsigned short* __restrict__ outH) {
  const int row = blockIdx.x;
  const float* yr = y + (size_t)row * kH;
  const float* rr = resid + (size_t)row * kH;
  float v[4];
  float s = 0.f, sq = 0.f;
#pragma unroll
  for (int i = 0; i < 4; ++i) {
    const int c = threadIdx.x + i * 256;
    const float t = yr[c] + rr[c];
    v[i] = t;
    s += t;
    sq += t * t;
  }
#pragma unroll
  for (int m = 1; m < 32; m <<= 1) {
    s += __shfl_xor(s, m, 32);
    sq += __shfl_xor(sq, m, 32);
  }
  __shared__ float rs[8], rq[8];
  if ((threadIdx.x & 31) == 0) { rs[threadIdx.x >> 5] = s; rq[threadIdx.x >> 5] = sq; }
  __syncthreads();
  float ts = 0.f, tq = 0.f;
#pragma unroll
  for (int w = 0; w < 8; ++w) { ts += rs[w]; tq += rq[w]; }
  const float mean = ts * (1.f / kH);
  const float var  = tq * (1.f / kH) - mean * mean;
  const float inv  = rsqrtf(var + 1e-5f);
#pragma unroll
  for (int i = 0; i < 4; ++i) {
    const int c = threadIdx.x + i * 256;
    const float ov = (v[i] - mean) * inv * g[c] + beta[c];
    outF[(size_t)row * kH + c] = ov;
    if (outH) outH[(size_t)row * kH + c] = f2bf(ov);
  }
}

// ---------------------------------------------------------------------------
extern "C" void kernel_launch(void* const* d_in, const int* in_sizes, int n_in,
                              void* d_out, int out_size, void* d_ws, size_t ws_size,
                              hipStream_t stream) {
  const float* x     = (const float*)d_in[0];
  const float* in_w  = (const float*)d_in[1];
  const float* in_b  = (const float*)d_in[2];
  const float* out_w = (const float*)d_in[3];
  const float* out_b = (const float*)d_in[4];
  const float* fc1_w = (const float*)d_in[5];
  const float* fc1_b = (const float*)d_in[6];
  const float* fc2_w = (const float*)d_in[7];
  const float* fc2_b = (const float*)d_in[8];
  const float* g1    = (const float*)d_in[9];
  const float* b1    = (const float*)d_in[10];
  const float* g2    = (const float*)d_in[11];
  const float* b2    = (const float*)d_in[12];
  float* out = (float*)d_out;

  char* ws = (char*)d_ws;
  size_t off = 0;
  auto alloc = [&](size_t bytes) -> char* {
    char* p = ws + off;
    off += (bytes + 255) & ~(size_t)255;
    return p;
  };

  unsigned short* xh    = (unsigned short*)alloc((size_t)kBS * kH * 2);
  unsigned short* wInH  = (unsigned short*)alloc((size_t)3 * kH * kH * 2);
  unsigned short* wOutH = (unsigned short*)alloc((size_t)kH * kH * 2);
  unsigned short* wF1H  = (unsigned short*)alloc((size_t)kH * kH * 2);
  unsigned short* wF2H  = (unsigned short*)alloc((size_t)kH * kH * 2);
  float*          maskb = (float*)alloc((size_t)kBS * 4);
  unsigned short* qkvh  = (unsigned short*)alloc((size_t)kBS * 3 * kH * 2);
  unsigned short* qh    = (unsigned short*)alloc((size_t)kBS * kH * 2);
  unsigned short* khp   = (unsigned short*)alloc((size_t)kBS * kH * 2);
  unsigned short* vth   = (unsigned short*)alloc((size_t)kBS * kH * 2);
  unsigned short* ctxh  = (unsigned short*)alloc((size_t)kBS * kH * 2);
  float*          x1F   = (float*)alloc((size_t)kBS * kH * 4);
  unsigned short* x1H   = (unsigned short*)alloc((size_t)kBS * kH * 2);
  unsigned short* ffH   = (unsigned short*)alloc((size_t)kBS * kH * 2);
  float* attnF = (float*)qkvh;  // reuse: dead after scatter
  float* ff2F  = (float*)qh;    // reuse: qh+kh dead after attention
  (void)ws_size; (void)in_sizes; (void)n_in; (void)out_size;

  cvt_kernel<<<2048, 256, 0, stream>>>(x, xh, (size_t)kBS * kH);
  cvt_kernel<<<2048, 256, 0, stream>>>(in_w, wInH, (size_t)3 * kH * kH);
  cvt_kernel<<<1024, 256, 0, stream>>>(out_w, wOutH, (size_t)kH * kH);
  cvt_kernel<<<1024, 256, 0, stream>>>(fc1_w, wF1H, (size_t)kH * kH);
  cvt_kernel<<<1024, 256, 0, stream>>>(fc2_w, wF2H, (size_t)kH * kH);
  mask_kernel<<<kBS, 256, 0, stream>>>(x, maskb);

  gemm_bf16_kernel<1><<<dim3(3 * kH / 64, kBS / 128), 256, 0, stream>>>(
      xh, wInH, in_b, nullptr, qkvh, kBS, 3 * kH, kH);

  scatter_qkv_kernel<<<4096, 256, 0, stream>>>(qkvh, qh, khp, vth);

  attn_kernel<<<dim3(kB * kNH, kS / 128), 256, 0, stream>>>(qh, khp, vth, maskb,
                                                            ctxh);

  gemm_bf16_kernel<0><<<dim3(kH / 64, kBS / 128), 256, 0, stream>>>(
      ctxh, wOutH, out_b, attnF, nullptr, kBS, kH, kH);

  add_ln_kernel<<<kBS, 256, 0, stream>>>(attnF, x, g1, b1, x1F, x1H);

  gemm_bf16_kernel<2><<<dim3(kH / 64, kBS / 128), 256, 0, stream>>>(
      x1H, wF1H, fc1_b, nullptr, ffH, kBS, kH, kH);

  gemm_bf16_kernel<0><<<dim3(kH / 64, kBS / 128), 256, 0, stream>>>(
      ffH, wF2H, fc2_b, ff2F, nullptr, kBS, kH, kH);

  add_ln_kernel<<<kBS, 256, 0, stream>>>(ff2F, x1F, g2, b2, out, nullptr);
}
